// Quant_VoiceKeyModel_89979564851718
// MI455X (gfx1250) — matrix operand order repaired and checked by reading, not compiled, not run on hardware
//
#include <hip/hip_runtime.h>

// ---------------------------------------------------------------------------
// VoiceKey fused kernel for gfx1250 (MI455X).
// Full-row conv reformulation + f16 WMMA (f32 accum) for all heavy GEMMs.
// ---------------------------------------------------------------------------

typedef __attribute__((ext_vector_type(16))) _Float16 v16h;
typedef __attribute__((ext_vector_type(8)))  float    v8f;

#define EPS 1e-5f

// Workspace layout (units of _Float16)
#define N_W2    (26*32*16)   // w2 as [tap][och][ich]
#define N_W3    (25*64*32)   // w3 as [tap][och][ich]
#define N_WQ    (8*64*64)    // [h][e][d]
#define N_WFC   (64*512)     // [o][k]
#define OFF_W2  0
#define OFF_W3  (OFF_W2 + N_W2)
#define OFF_WQ  (OFF_W3 + N_W3)
#define OFF_WK  (OFF_WQ + N_WQ)
#define OFF_WV  (OFF_WK + N_WQ)
#define OFF_WFC (OFF_WV + N_WQ)
#define N_PREP  (OFF_WFC + N_WFC)

union FragH { v16h v; unsigned u[8]; };

__device__ __forceinline__ v8f wmma16(v16h a, v16h b, v8f c) {
  return __builtin_amdgcn_wmma_f32_16x16x32_f16(false, a, false, b, (short)0, c,
                                                false, false);
}

// A-matrix fragment (16x32 f16): lane m = l&15, lane-half kh picks K groups.
// slots 0..7 -> K = 2s + 16*kh
__device__ __forceinline__ v16h ldA(const _Float16* base, int kh) {
  FragH f;
#pragma unroll
  for (int s = 0; s < 8; ++s) f.u[s] = *(const unsigned*)(base + 2*s + 16*kh);
  return f.v;
}
// A fragment whose K<16 / K>=16 halves come from two different rows (tap pair)
__device__ __forceinline__ v16h ldA2(const _Float16* b0, const _Float16* b1, int kh) {
  FragH f;
  const _Float16* bp = kh ? b1 : b0;
#pragma unroll
  for (int s = 0; s < 8; ++s) f.u[s] = *(const unsigned*)(bp + 2*s);
  return f.v;
}
// B-matrix fragment (32x16 f16): lane n = l&15; K = 2s + 16*kh.
// Caller folds "+16*kh" into base -> 8 consecutive dwords.
__device__ __forceinline__ v16h ldB(const _Float16* base) {
  FragH f;
#pragma unroll
  for (int s = 0; s < 8; ++s) f.u[s] = *(const unsigned*)(base + 2*s);
  return f.v;
}

// swish via single-instruction v_rcp_f32 (result is rounded to f16 anyway)
__device__ __forceinline__ float swish(float y) {
  return y * __builtin_amdgcn_rcpf(1.f + __expf(-y));
}

// ---------------------------------------------------------------------------
// Weight prep: convert / re-layout weights to f16 in workspace (L2-resident).
// ---------------------------------------------------------------------------
__global__ void vk_prep(const float* __restrict__ w2, const float* __restrict__ w3,
                        const float* __restrict__ Wq, const float* __restrict__ Wk,
                        const float* __restrict__ Wv, const float* __restrict__ Wfc,
                        _Float16* __restrict__ ws) {
  int i = blockIdx.x * blockDim.x + threadIdx.x;
  if (i >= N_PREP) return;
  if (i < OFF_W3) {                    // w2 (32,16,26) -> [t][och][ich]
    int j = i - OFF_W2;
    int t = j / 512, rem = j % 512, och = rem / 16, ich = rem % 16;
    ws[i] = (_Float16)w2[(och*16 + ich)*26 + t];
  } else if (i < OFF_WQ) {             // w3 (64,32,25) -> [t][och][ich]
    int j = i - OFF_W3;
    int t = j / 2048, rem = j % 2048, och = rem / 32, ich = rem % 32;
    ws[i] = (_Float16)w3[(och*32 + ich)*25 + t];
  } else if (i < OFF_WK)  { ws[i] = (_Float16)Wq[i - OFF_WQ]; }
  else if (i < OFF_WV)    { ws[i] = (_Float16)Wk[i - OFF_WK]; }
  else if (i < OFF_WFC)   { ws[i] = (_Float16)Wv[i - OFF_WV]; }
  else                    { ws[i] = (_Float16)Wfc[i - OFF_WFC]; }
}

// ---------------------------------------------------------------------------
// Fused per-row kernel: encoder (conv1 VALU, conv2/conv3 WMMA) + MHA (WMMA)
// grid = (B, 2), block = 128 (4 wave32)
// ---------------------------------------------------------------------------
__global__ __launch_bounds__(128)
void vk_main(const float* __restrict__ audio1, const float* __restrict__ audio2,
             const float* __restrict__ w1,
             const float* __restrict__ b1, const float* __restrict__ g1,
             const float* __restrict__ be1, const float* __restrict__ m1,
             const float* __restrict__ v1,
             const float* __restrict__ b2, const float* __restrict__ g2,
             const float* __restrict__ be2, const float* __restrict__ m2,
             const float* __restrict__ v2,
             const float* __restrict__ b3, const float* __restrict__ g3,
             const float* __restrict__ be3, const float* __restrict__ m3,
             const float* __restrict__ v3,
             const float* __restrict__ bq, const float* __restrict__ bk,
             const float* __restrict__ bv, const float* __restrict__ bfc,
             const float* __restrict__ Wout, const float* __restrict__ bout,
             const _Float16* __restrict__ ws, float* __restrict__ out)
{
  __shared__ _Float16 featL[32*64];                  // [s][d], rows 22..31 = 0
  __shared__ float sc1[16], sh1[16], sc2[32], sh2[32], sc3[64], sh3[64];
  __shared__ float w1s[16*26];
  __shared__ float meanv[64];
  __shared__ float pb[128];
  __shared__ __align__(16) union {
    struct {                                         // encoder phase
      float    xrow[600];
      _Float16 y1[592*16];                           // [pos][ch], pad zeroed
      _Float16 y2[560*32];                           // [pos][ch]
    } a;
    struct {                                         // attention phase
      _Float16 Qh[32*64];                            // [s][e]
      _Float16 Kh[32*64];                            // [t][e]
      _Float16 VhT[64*32];                           // [e][t]
      float    Sh[32*32];                            // scores f32
      _Float16 Ah[32*32];                            // softmax(A), masked
      _Float16 OL[32*512];                           // [s][h*64+e]
    } b;
  } u;

  const int tid   = threadIdx.x;
  const int lane  = tid & 31;
  const int wave  = tid >> 5;
  const int l15   = lane & 15;
  const int kh    = lane >> 4;
  const int row   = blockIdx.x;
  const int which = blockIdx.y;
  const int NB    = gridDim.x;
  const float* audio = which ? audio2 : audio1;

  // ---- setup: fused conv-bias+BN scale/shift, w1 and audio row to LDS ----
  if (tid < 16) {
    float s = g1[tid] * rsqrtf(v1[tid] + EPS);
    sc1[tid] = s; sh1[tid] = (b1[tid] - m1[tid]) * s + be1[tid];
  } else if (tid < 48) {
    int c = tid - 16;
    float s = g2[c] * rsqrtf(v2[c] + EPS);
    sc2[c] = s; sh2[c] = (b2[c] - m2[c]) * s + be2[c];
  } else if (tid < 112) {
    int c = tid - 48;
    float s = g3[c] * rsqrtf(v3[c] + EPS);
    sc3[c] = s; sh3[c] = (b3[c] - m3[c]) * s + be3[c];
  }
  for (int i = tid; i < 16*26; i += 128) w1s[i] = w1[i];
  for (int i = tid; i < 600;   i += 128) u.a.xrow[i] = audio[row*600 + i];
  __syncthreads();

  // ---- conv1 (VALU f32, 3% of FLOPs): 575 positions x 16 ch ----
  for (int i = tid; i < 575*16; i += 128) {
    int pos = i >> 4, ch = i & 15;
    float acc = 0.f;
#pragma unroll
    for (int k = 0; k < 26; ++k) acc += w1s[ch*26 + k] * u.a.xrow[pos + k];
    u.a.y1[pos*16 + ch] = (_Float16)swish(acc * sc1[ch] + sh1[ch]);
  }
  for (int i = 575*16 + tid; i < 592*16; i += 128) u.a.y1[i] = (_Float16)0.f;
  __syncthreads();

  // ---- conv2 (WMMA): M=32 och, N=550 pos (35 tiles), K=13 tap-pairs x 32 ----
  const _Float16* W2c = ws + OFF_W2;
  for (int nt = wave; nt < 35; nt += 4) {
    int p = nt*16 + l15;                             // this lane's column (pos)
    v8f acc0 = {}; v8f acc1 = {};
#pragma unroll 1
    for (int kp = 0; kp < 13; ++kp) {
      int t = 2*kp;
      v16h bf = ldB(u.a.y1 + (p + t + kh)*16);       // K<16 -> tap t, K>=16 -> t+1
      v16h a0 = ldA2(W2c + (t*32 + l15)*16,      W2c + ((t+1)*32 + l15)*16,      kh);
      v16h a1 = ldA2(W2c + (t*32 + 16 + l15)*16, W2c + ((t+1)*32 + 16 + l15)*16, kh);
      acc0 = wmma16(a0, bf, acc0);
      acc1 = wmma16(a1, bf, acc1);
    }
#pragma unroll
    for (int r = 0; r < 8; ++r) {
      int o0 = 8*kh + r, o1 = 16 + 8*kh + r;
      u.a.y2[p*32 + o0] = (_Float16)swish(acc0[r]*sc2[o0] + sh2[o0]);
      u.a.y2[p*32 + o1] = (_Float16)swish(acc1[r]*sc2[o1] + sh2[o1]);
    }
  }
  __syncthreads();

  // ---- conv3 (WMMA): M=64 och (wave=mtile), N=22 chunks, K=25 taps x 32ch ----
  const _Float16* W3c = ws + OFF_W3;
  {
    int och = wave*16 + l15;
    int c0  = l15;
    int c1t = 16 + l15;
    int c1  = c1t < 22 ? c1t : 21;                   // clamp pad columns
    v8f acc0 = {}; v8f acc1 = {};
#pragma unroll 1
    for (int t = 0; t < 25; ++t) {
      v16h a  = ldA(W3c + (t*64 + och)*32, kh);
      v16h b0 = ldB(u.a.y2 + (25*c0 + t)*32 + 16*kh);
      v16h b1 = ldB(u.a.y2 + (25*c1 + t)*32 + 16*kh);
      acc0 = wmma16(a, b0, acc0);
      acc1 = wmma16(a, b1, acc1);
    }
#pragma unroll
    for (int r = 0; r < 8; ++r) {
      int oc = wave*16 + 8*kh + r;
      featL[c0*64 + oc] = (_Float16)swish(acc0[r]*sc3[oc] + sh3[oc]);
      float y = swish(acc1[r]*sc3[oc] + sh3[oc]);
      featL[c1t*64 + oc] = (_Float16)(c1t < 22 ? y : 0.f);
    }
  }
  __syncthreads();

  // ---- MHA: loop heads, 4 waves cooperate within each head ----
  const _Float16* Wqc  = ws + OFF_WQ;
  const _Float16* Wkc  = ws + OFF_WK;
  const _Float16* Wvc  = ws + OFF_WV;
  const _Float16* Wfcc = ws + OFF_WFC;

  for (int h = 0; h < 8; ++h) {
    {   // Q,K,V projections; wave handles e-tile = wave
      int e = wave*16 + l15;
      v8f q0={},q1={},k0={},k1={},vv0={},vv1={};
#pragma unroll 1
      for (int kt = 0; kt < 2; ++kt) {
        v16h a0  = ldA(featL + l15*64 + kt*32, kh);
        v16h a1  = ldA(featL + (16 + l15)*64 + kt*32, kh);
        v16h bq_ = ldB(Wqc + (h*64 + e)*64 + kt*32 + 16*kh);
        v16h bk_ = ldB(Wkc + (h*64 + e)*64 + kt*32 + 16*kh);
        v16h bv_ = ldB(Wvc + (h*64 + e)*64 + kt*32 + 16*kh);
        q0  = wmma16(a0, bq_, q0);  q1  = wmma16(a1, bq_, q1);
        k0  = wmma16(a0, bk_, k0);  k1  = wmma16(a1, bk_, k1);
        vv0 = wmma16(a0, bv_, vv0); vv1 = wmma16(a1, bv_, vv1);
      }
      float bqv = bq[h*64 + e], bkv = bk[h*64 + e], bvv = bv[h*64 + e];
#pragma unroll
      for (int r = 0; r < 8; ++r) {
        int s0 = 8*kh + r, s1 = 16 + 8*kh + r;
        u.b.Qh[s0*64 + e]  = (_Float16)(q0[r] + bqv);
        u.b.Qh[s1*64 + e]  = (_Float16)(q1[r] + bqv);
        u.b.Kh[s0*64 + e]  = (_Float16)(k0[r] + bkv);
        u.b.Kh[s1*64 + e]  = (_Float16)(k1[r] + bkv);
        u.b.VhT[e*32 + s0] = (_Float16)(vv0[r] + bvv);   // transposed [e][t]
        u.b.VhT[e*32 + s1] = (_Float16)(vv1[r] + bvv);
      }
    }
    __syncthreads();

    {   // scores S = Q K^T ; wave -> (mt,nt) quadrant
      int mt = wave >> 1, ntt = wave & 1;
      v8f sa = {};
#pragma unroll 1
      for (int kt = 0; kt < 2; ++kt) {
        v16h a  = ldA(u.b.Qh + (mt*16 + l15)*64 + kt*32, kh);
        v16h bb = ldB(u.b.Kh + (ntt*16 + l15)*64 + kt*32 + 16*kh);
        sa = wmma16(a, bb, sa);
      }
#pragma unroll
      for (int r = 0; r < 8; ++r)
        u.b.Sh[(mt*16 + 8*kh + r)*32 + ntt*16 + l15] = sa[r];
    }
    __syncthreads();

    // softmax over valid t=0..21; pad rows/cols zeroed (masks pad V rows)
    if (tid < 22) {
      float mx = -3.0e38f;
      for (int t = 0; t < 22; ++t) mx = fmaxf(mx, u.b.Sh[tid*32 + t]);
      float sum = 0.f;
      for (int t = 0; t < 22; ++t) sum += __expf(u.b.Sh[tid*32 + t] - mx);
      float inv = __builtin_amdgcn_rcpf(sum);
      for (int t = 0; t < 22; ++t)
        u.b.Ah[tid*32 + t] = (_Float16)(__expf(u.b.Sh[tid*32 + t] - mx) * inv);
      for (int t = 22; t < 32; ++t) u.b.Ah[tid*32 + t] = (_Float16)0.f;
    } else if (tid < 32) {
      for (int t = 0; t < 32; ++t) u.b.Ah[tid*32 + t] = (_Float16)0.f;
    }
    __syncthreads();

    {   // O = A V ; wave -> e-tile; K = t (single 32-K tile)
      int e = wave*16 + l15;
      v16h bb = ldB(u.b.VhT + e*32 + 16*kh);
      v16h a0 = ldA(u.b.Ah + l15*32, kh);
      v16h a1 = ldA(u.b.Ah + (16 + l15)*32, kh);
      v8f o0 = {}; v8f o1 = {};
      o0 = wmma16(a0, bb, o0);
      o1 = wmma16(a1, bb, o1);
#pragma unroll
      for (int r = 0; r < 8; ++r) {
        int s0 = 8*kh + r, s1 = 16 + 8*kh + r;
        u.b.OL[s0*512 + h*64 + e] = (_Float16)o0[r];
        u.b.OL[s1*512 + h*64 + e] = (_Float16)o1[r];
      }
    }
    __syncthreads();
  }

  // ---- fc (K=512) + masked mean over s<22 ----
  {
    int o = wave*16 + l15;
    v8f f0 = {}; v8f f1 = {};
#pragma unroll 1
    for (int kt = 0; kt < 16; ++kt) {
      v16h a0 = ldA(u.b.OL + l15*512 + kt*32, kh);
      v16h a1 = ldA(u.b.OL + (16 + l15)*512 + kt*32, kh);
      v16h bb = ldB(Wfcc + o*512 + kt*32 + 16*kh);
      f0 = wmma16(a0, bb, f0);
      f1 = wmma16(a1, bb, f1);
    }
    float part = 0.f;
#pragma unroll
    for (int r = 0; r < 8; ++r) {
      if (8*kh + r < 22)      part += f0[r];
      if (16 + 8*kh + r < 22) part += f1[r];
    }
    pb[tid] = part;
  }
  __syncthreads();
  if (tid < 64) {
    int w = tid >> 4, l_ = tid & 15;
    float sum = pb[w*32 + l_] + pb[w*32 + l_ + 16];
    meanv[tid] = sum * (1.f/22.f) + bfc[tid];
  }
  __syncthreads();
  // ---- final 64x64 linear (VALU) ----
  if (tid < 64) {
    float r = bout[tid];
#pragma unroll 1
    for (int o = 0; o < 64; ++o) r += meanv[o] * Wout[tid*64 + o];
    out[((size_t)which * NB + row)*64 + tid] = r;
  }
}

// ---------------------------------------------------------------------------
extern "C" void kernel_launch(void* const* d_in, const int* in_sizes, int n_in,
                              void* d_out, int out_size, void* d_ws, size_t ws_size,
                              hipStream_t stream) {
  (void)n_in; (void)out_size; (void)ws_size;
  const float* audio1 = (const float*)d_in[0];
  const float* audio2 = (const float*)d_in[1];
  const float* w1  = (const float*)d_in[2];
  const float* b1  = (const float*)d_in[3];
  const float* g1  = (const float*)d_in[4];
  const float* be1 = (const float*)d_in[5];
  const float* m1  = (const float*)d_in[6];
  const float* v1  = (const float*)d_in[7];
  const float* w2  = (const float*)d_in[8];
  const float* b2  = (const float*)d_in[9];
  const float* g2  = (const float*)d_in[10];
  const float* be2 = (const float*)d_in[11];
  const float* m2  = (const float*)d_in[12];
  const float* v2  = (const float*)d_in[13];
  const float* w3  = (const float*)d_in[14];
  const float* b3  = (const float*)d_in[15];
  const float* g3  = (const float*)d_in[16];
  const float* be3 = (const float*)d_in[17];
  const float* m3  = (const float*)d_in[18];
  const float* v3  = (const float*)d_in[19];
  const float* Wq  = (const float*)d_in[20];
  const float* bq  = (const float*)d_in[21];
  const float* Wk  = (const float*)d_in[22];
  const float* bk  = (const float*)d_in[23];
  const float* Wv  = (const float*)d_in[24];
  const float* bv  = (const float*)d_in[25];
  const float* Wfc = (const float*)d_in[26];
  const float* bfc = (const float*)d_in[27];
  const float* Wout= (const float*)d_in[28];
  const float* bout= (const float*)d_in[29];
  _Float16* ws = (_Float16*)d_ws;
  float* out = (float*)d_out;

  int NB = in_sizes[0] / 600;                 // batch (4096)

  vk_prep<<<(N_PREP + 255)/256, 256, 0, stream>>>(w2, w3, Wq, Wk, Wv, Wfc, ws);
  dim3 grid(NB, 2);
  vk_main<<<grid, 128, 0, stream>>>(audio1, audio2, w1,
                                    b1, g1, be1, m1, v1,
                                    b2, g2, be2, m2, v2,
                                    b3, g3, be3, m3, v3,
                                    bq, bk, bv, bfc, Wout, bout, ws, out);
}
